// GaussianRenderer_67242007986512
// MI455X (gfx1250) — matrix operand order repaired and checked by reading, not compiled, not run on hardware
//
#include <hip/hip_runtime.h>
#include <hip/hip_bf16.h>

// ---------------------------------------------------------------------------
// Gaussian splat renderer for MI455X (gfx1250, wave32).
// Chunk reduction einsum('khw,kc->chw') + a_sum mapped onto
// V_WMMA_F32_16X16X32_F16 with:
//   A = channel matrix (16 x 32 f16): rows 0..3 = {R, G, B, 1.0}, rows 4..15 = 0
//       staged in LDS in the exact A-fragment layout -> 2x ds_load_b128/lane
//   B = alpha   matrix (32 x 16 f16): column n = pixel n of the wave's tile
//   D = f32 16x16: lane n (n<16) holds d[0..3] = {Rsum, Gsum, Bsum, a_sum}
//       for its own pixel -> extraction is lane-local, branch-free.
// ---------------------------------------------------------------------------

typedef __attribute__((ext_vector_type(16))) _Float16 v16h;
typedef __attribute__((ext_vector_type(8)))  _Float16 v8h;
typedef __attribute__((ext_vector_type(8)))  float    v8f;

#define WW 512
#define HH 512
#define HWPIX (WW * HH)
#define SH_C0 0.28209479177387814f
#define NLOG2E_HALF (-0.7213475204444817f)   /* -0.5 * log2(e) */

// ---------------------------------------------------------------- phase 1a --
__global__ void gs_prep(const float* __restrict__ pos,
                        const float* __restrict__ scales,
                        const float* __restrict__ opac,
                        const float* __restrict__ sh,
                        const float* __restrict__ view,
                        const float* __restrict__ proj,
                        float*  __restrict__ keys,
                        float4* __restrict__ gtmp,
                        float4* __restrict__ ctmp,
                        int N)
{
    int i = blockIdx.x * blockDim.x + threadIdx.x;
    if (i >= N) return;

    float p0 = pos[3*i+0], p1 = pos[3*i+1], p2 = pos[3*i+2];

    // pos_cam = pos_h @ view.T  -> depth = row 2 of view
    float depth = view[8]*p0 + view[9]*p1 + view[10]*p2 + view[11];
    bool valid = depth > 0.1f;

    // M = proj @ view (rows 0, 1, 3 needed)
    float m0[4], m1[4], m3[4];
#pragma unroll
    for (int k = 0; k < 4; ++k) {
        m0[k] = proj[0]*view[k]  + proj[1]*view[4+k]  + proj[2]*view[8+k]  + proj[3]*view[12+k];
        m1[k] = proj[4]*view[k]  + proj[5]*view[4+k]  + proj[6]*view[8+k]  + proj[7]*view[12+k];
        m3[k] = proj[12]*view[k] + proj[13]*view[4+k] + proj[14]*view[8+k] + proj[15]*view[12+k];
    }
    float pp0 = m0[0]*p0 + m0[1]*p1 + m0[2]*p2 + m0[3];
    float pp1 = m1[0]*p0 + m1[1]*p1 + m1[2]*p2 + m1[3];
    float pp3 = m3[0]*p0 + m3[1]*p1 + m3[2]*p2 + m3[3];
    float inv_w = 1.0f / (pp3 + 1e-8f);
    float ndc0 = pp0 * inv_w;
    float ndc1 = pp1 * inv_w;

    // astype(int32) truncates toward zero; clamp to stay in int range
    float sx = truncf(fminf(fmaxf((ndc0 + 1.0f) * 0.5f * (float)WW, -1e9f), 1e9f));
    float sy = truncf(fminf(fmaxf((1.0f - ndc1) * 0.5f * (float)HH, -1e9f), 1e9f));

    float rmax = fmaxf(fmaxf(scales[3*i+0], scales[3*i+1]), scales[3*i+2]);
    float rp   = fmaxf(rmax * 2.0f * ((float)WW * 0.25f), 1.0f);

    float oe = opac[i] * (valid ? 1.0f : 0.0f);   // fold valid mask into opacity

    float cr = fminf(fmaxf(sh[48*i + 0] * SH_C0 + 0.5f, 0.0f), 1.0f);
    float cg = fminf(fmaxf(sh[48*i + 1] * SH_C0 + 0.5f, 0.0f), 1.0f);
    float cb = fminf(fmaxf(sh[48*i + 2] * SH_C0 + 0.5f, 0.0f), 1.0f);

    keys[i] = valid ? depth : __builtin_inff();
    gtmp[i] = make_float4(sx, sy, 1.0f / rp, oe);
    ctmp[i] = make_float4(cr, cg, cb, 1.0f);      // .w = 1.0 -> a_sum row of A
}

// ---------------------------------------------------------------- phase 1b --
// Stable ascending sort by rank counting (O(N^2), N=4096 -> trivial).
__global__ void gs_sort(const float*  __restrict__ keys,
                        const float4* __restrict__ gtmp,
                        const float4* __restrict__ ctmp,
                        float4* __restrict__ gsorted,
                        float4* __restrict__ csorted,
                        int N)
{
    int i = blockIdx.x * blockDim.x + threadIdx.x;
    if (i >= N) return;
    float ki = keys[i];
    int rank = 0;
    for (int j = 0; j < N; ++j) {
        float kj = keys[j];
        rank += (kj < ki) || (kj == ki && j < i);   // stable tie-break = argsort
    }
    gsorted[rank] = gtmp[i];
    csorted[rank] = ctmp[i];
}

// ---------------------------------------------------------------- phase 2 ---
// One wave32 renders 16 consecutive pixels; loops over all chunks of 32.
__global__ __launch_bounds__(256) void gs_render(const float4* __restrict__ gsrt,
                                                 const float4* __restrict__ csrt,
                                                 float* __restrict__ out,
                                                 int nch)
{
    __shared__ float4 sg[32];                       // {cx, cy, 1/r, o*valid}
    __shared__ __align__(16) _Float16 scT[16 * 32]; // A fragment: [row][k]

    const int tid  = threadIdx.x;
    const int lane = tid & 31;
    const int m    = lane & 15;            // pixel slot / A-row (channel) index
    const int g    = lane >> 4;            // lane half: selects K halves
    const int tile = blockIdx.x * (blockDim.x >> 5) + (tid >> 5);
    const int pixBase = tile << 4;         // 16 pixels per wave
    const int px = pixBase + m;            // this lane's pixel (B column m)
    const float fx = (float)(px & (WW - 1));
    const float fy = (float)(px >> 9);

    // zero A rows 4..15 once (halves 128..511 = dwords 64..255)
    if (tid < 192) ((float*)scT)[64 + tid] = 0.0f;

    float crA = 0.0f, cgA = 0.0f, cbA = 0.0f, aacc = 0.0f;

    for (int ch = 0; ch < nch; ++ch) {
        __syncthreads();
        if (tid < 32) {                     // wave-uniform branch (wave 0 only)
            sg[tid] = gsrt[(ch << 5) + tid];
            float4 C = csrt[(ch << 5) + tid];
            scT[0 * 32 + tid] = (_Float16)C.x;   // row 0 = R
            scT[1 * 32 + tid] = (_Float16)C.y;   // row 1 = G
            scT[2 * 32 + tid] = (_Float16)C.z;   // row 2 = B
            scT[3 * 32 + tid] = (_Float16)C.w;   // row 3 = 1.0 -> a_sum
        }
        __syncthreads();

        // prefetch next chunk into cache (global_prefetch_b8); uniform address
        int nxt = (ch + 1 < nch) ? (ch + 1) : ch;
        __builtin_prefetch(&gsrt[(nxt << 5)], 0, 1);
        __builtin_prefetch(&csrt[(nxt << 5)], 0, 1);

        // A (16x32 f16): lane L -> row m; element e -> K = 16g+e.
        // Row m is contiguous -> two ds_load_b128.
        const v8h* rowp = (const v8h*)&scT[m * 32];
        v8h alo = rowp[2 * g];      // halves 16g .. 16g+7     (K = 16g+e)
        v8h ahi = rowp[2 * g + 1];  // halves 16g+8 .. 16g+15  (K = 16g+8+e)
        v16h a;
#pragma unroll
        for (int e = 0; e < 8; ++e) { a[e] = alo[e]; a[8 + e] = ahi[e]; }

        // B (32x16 f16) = alphas: lane L -> column N = L%16 (its pixel);
        //   element e -> K = 16*(L/16) + e.  Each lane: 16 exp evals.
        v16h b;
#pragma unroll
        for (int e = 0; e < 16; ++e) {
            float4 G = sg[(g << 4) + e];
            float dx = (fx - G.x) * G.z;
            float dy = (fy - G.y) * G.z;
            float s  = dx * dx + dy * dy;
            float w  = __builtin_amdgcn_exp2f(NLOG2E_HALF * s);  // exp(-0.5 s)
            float ins = (fabsf(dx) < 3.0f && fabsf(dy) < 3.0f) ? 1.0f : 0.0f;
            b[e] = (_Float16)fminf(G.w * w * ins, 0.99f);        // a in [0, .99]
        }

        v8f cz = {};
        // D = A x B (f32 accum). Lanes 0-15: d[0..3] = {Rsum,Gsum,Bsum,a_sum}
        // for pixel px. Lanes 16-31: rows 8-15 of D = 0 (A rows 4-15 are 0).
        v8f d = __builtin_amdgcn_wmma_f32_16x16x32_f16(
            /*neg_a=*/false, a, /*neg_b=*/false, b,
            /*c_mod=*/(short)0, cz, /*reuse_a=*/false, /*reuse_b=*/false);

        // Branch-free, lane-local accumulation (upper half adds exact zeros:
        // 0 * rcp == 0). EXEC stays all-1s for the next WMMA.
        float rcp = __builtin_amdgcn_rcpf(d[3] + 1e-8f);  // v_rcp_f32
        crA  += d[0] * rcp;
        cgA  += d[1] * rcp;
        cbA  += d[2] * rcp;
        aacc += fminf(d[3], 0.99f);
    }

    // finalize: color = 1 (bg init) + sum + 1*max(1-acc, 0);  out ch3 = acc
    if (g == 0) {
        float bgterm = 1.0f + fmaxf(1.0f - aacc, 0.0f);
        out[0 * HWPIX + px] = bgterm + crA;
        out[1 * HWPIX + px] = bgterm + cgA;
        out[2 * HWPIX + px] = bgterm + cbA;
        out[3 * HWPIX + px] = aacc;
    }
}

// ---------------------------------------------------------------------------
extern "C" void kernel_launch(void* const* d_in, const int* in_sizes, int n_in,
                              void* d_out, int out_size, void* d_ws, size_t ws_size,
                              hipStream_t stream)
{
    const float* positions = (const float*)d_in[0];
    // d_in[1] = rotations (unused by the reference)
    const float* scales    = (const float*)d_in[2];
    const float* opac      = (const float*)d_in[3];
    const float* sh        = (const float*)d_in[4];
    const float* view      = (const float*)d_in[5];
    const float* proj      = (const float*)d_in[6];
    float* out = (float*)d_out;

    const int N   = in_sizes[3];        // opacities: (N,1)
    const int nch = N / 32;

    // workspace layout (16B aligned)
    char* ws = (char*)d_ws;
    size_t off = 0;
    float*  keys = (float*)(ws + off);  off += ((size_t)N * 4 + 15) & ~(size_t)15;
    float4* gtmp = (float4*)(ws + off); off += (size_t)N * 16;
    float4* ctmp = (float4*)(ws + off); off += (size_t)N * 16;
    float4* gsrt = (float4*)(ws + off); off += (size_t)N * 16;
    float4* csrt = (float4*)(ws + off); off += (size_t)N * 16;

    int pb = (N + 255) / 256;
    gs_prep<<<pb, 256, 0, stream>>>(positions, scales, opac, sh, view, proj,
                                    keys, gtmp, ctmp, N);
    gs_sort<<<pb, 256, 0, stream>>>(keys, gtmp, ctmp, gsrt, csrt, N);

    // 262144 pixels / (16 px/wave * 8 waves/block) = 2048 blocks
    gs_render<<<HWPIX / (16 * 8), 256, 0, stream>>>(gsrt, csrt, out, nch);
}
